// NCA_33655363731628
// MI455X (gfx1250) — matrix operand order repaired and checked by reading, not compiled, not run on hardware
//
#include <hip/hip_runtime.h>

// ---------------- problem constants ----------------
#define BATCH 16
#define IMH   256
#define IMW   256
#define NCH   16      // channels C
#define HID   64
#define TS    16      // spatial tile side (16x16 pixels per block)
#define NPIX  (BATCH * IMH * IMW)

// LDS strides (in _Float16 elements) chosen to avoid bank conflicts
#define PSTR  40      // perception row stride (32 used + pad), 80B -> conflict-free b128
#define HSTR  72      // hidden row stride (64 used + pad), 144B -> conflict-free b128

typedef __attribute__((ext_vector_type(16))) _Float16 v16h;
typedef __attribute__((ext_vector_type(8)))  _Float16 v8h;
typedef __attribute__((ext_vector_type(8)))  float    v8f;
typedef __attribute__((ext_vector_type(4)))  float    v4f;

__device__ __forceinline__ v8f wmma16(v16h a, v16h b, v8f c) {
  // D(16x16 f32) = A(16x32 f16) * B(32x16 f16) + C
  return __builtin_amdgcn_wmma_f32_16x16x32_f16(false, a, false, b, (short)0, c,
                                                false, false);
}

__device__ __forceinline__ v16h cat8(v8h lo, v8h hi) {
  v16h r;
#pragma unroll
  for (int i = 0; i < 8; ++i) { r[i] = lo[i]; r[i + 8] = hi[i]; }
  return r;
}

// =====================================================================
// Kernel 1: perception + 3-layer MLP via WMMA; writes x_new to out and
// compact channel-3 planes (x, x_new) to workspace.
// =====================================================================
__global__ __launch_bounds__(256) void nca_mlp_kernel(
    const float* __restrict__ x,  const float* __restrict__ w1,
    const float* __restrict__ b1, const float* __restrict__ w2,
    const float* __restrict__ b2, const float* __restrict__ w3,
    float* __restrict__ out, float* __restrict__ x3p, float* __restrict__ xn3p)
{
  __shared__ float     sarr[18][20];          // channel-sum with halo
  __shared__ _Float16  perc[256 * PSTR];      // [pixel][32] A-matrix source
  __shared__ _Float16  wfrag[14 * 512];       // 14 B-fragments, [frag][lane][16]
  __shared__ _Float16  hbuf[8 * 16 * HSTR];   // per-wave hidden staging (h1 then h2)

  const int tid  = threadIdx.x;
  const int blk  = blockIdx.x;
  const int bimg = blk >> 8;                  // batch image
  const int tix  = blk & 255;
  const int th   = (tix >> 4) << 4;           // tile origin h
  const int tw   = (tix & 15) << 4;           // tile origin w

  // ---- Phase 0a: halo channel-sum s, and inner-pixel x -> perc (f16) ----
  for (int p = tid; p < 324; p += 256) {      // 18x18 positions
    const int r = p / 18, c = p % 18;
    const int gh = th + r - 1, gw = tw + c - 1;
    float s = 0.f;
    float xv[16];
    const bool inb = (gh >= 0) & (gh < IMH) & (gw >= 0) & (gw < IMW);
    if (inb) {
      const float* px = x + (((size_t)(bimg * IMH + gh) * IMW + gw) << 4);
#pragma unroll
      for (int i = 0; i < 4; ++i) {
        v4f v = *(const v4f*)(px + 4 * i);
        xv[4*i+0] = v.x; xv[4*i+1] = v.y; xv[4*i+2] = v.z; xv[4*i+3] = v.w;
        s += v.x + v.y + v.z + v.w;
      }
    } else {
#pragma unroll
      for (int i = 0; i < 16; ++i) xv[i] = 0.f;
    }
    sarr[r][c] = s;
    if (r >= 1 && r <= 16 && c >= 1 && c <= 16) {
      const int pl = ((r - 1) << 4) + (c - 1);
      unsigned* prow = (unsigned*)&perc[pl * PSTR];
#pragma unroll
      for (int i = 0; i < 8; ++i) {           // channels 0..15 (x), packed pairs
        union { _Float16 h[2]; unsigned u; } pk;
        pk.h[0] = (_Float16)xv[2 * i]; pk.h[1] = (_Float16)xv[2 * i + 1];
        prow[i] = pk.u;
      }
#pragma unroll
      for (int i = 9; i < 16; ++i) prow[i] = 0u;   // channels 18..31 = 0 (K pad)
    }
  }

  // ---- Phase 0b: swizzle weights into per-lane B fragments ----
  // B(32x16) layout: lane n (0..15) holds K = kc*32 + 0..15 (elems 0..15),
  // lane n+16 holds K = kc*32 + 16..31.  Frags: 0..3 L1 | 4..11 L2 | 12..13 L3.
  for (int idx = tid; idx < 14 * 512; idx += 256) {
    const int f = idx >> 9;
    const int rem = idx & 511;
    const int l = rem >> 4, e = rem & 15;
    const int khalf = l >> 4;                 // 0 or 1
    const int nl = l & 15;
    float val = 0.f;
    if (f < 4) {                              // layer 1 (effective K=18, padded)
      const int n = f * 16 + nl;
      const int k = khalf * 16 + e;
      if (k < 16) { val = w1[k * HID + n]; }
      else if (k == 16) { for (int j = 16; j < 32; ++j) val += w1[j * HID + n]; }
      else if (k == 17) { for (int j = 32; j < 48; ++j) val += w1[j * HID + n]; }
    } else if (f < 12) {                      // layer 2, f = 4 + kc*4 + tile
      const int t = (f - 4) & 3, kc = (f - 4) >> 2;
      const int n = t * 16 + nl;
      const int k = kc * 32 + khalf * 16 + e;
      val = w2[k * HID + n];
    } else {                                  // layer 3, f = 12 + kc
      const int kc = f - 12;
      const int k = kc * 32 + khalf * 16 + e;
      val = w3[k * NCH + nl];
    }
    wfrag[idx] = (_Float16)val;
  }
  __syncthreads();

  // ---- Phase 1: Sobel from s -> perc channels 16 (sx), 17 (sy) ----
  {
    const int r = (tid >> 4) + 1, c = (tid & 15) + 1;
    const float a00 = sarr[r-1][c-1], a01 = sarr[r-1][c], a02 = sarr[r-1][c+1];
    const float a10 = sarr[r  ][c-1],                     a12 = sarr[r  ][c+1];
    const float a20 = sarr[r+1][c-1], a21 = sarr[r+1][c], a22 = sarr[r+1][c+1];
    const float sx = (a02 - a00 + 2.f * (a12 - a10) + a22 - a20) * 0.125f;
    const float sy = (a20 - a00 + 2.f * (a21 - a01) + a22 - a02) * 0.125f;
    union { _Float16 h[2]; unsigned u; } pk;
    pk.h[0] = (_Float16)sx; pk.h[1] = (_Float16)sy;
    ((unsigned*)&perc[tid * PSTR])[8] = pk.u;
  }
  __syncthreads();

  // ---- Main compute: per wave, 2 groups of 16 pixels ----
  const int lane = tid & 31;
  const int wv   = tid >> 5;
  const int hi   = lane >> 4;                 // 0 / 1 half
  const int nl   = lane & 15;
  _Float16* hb   = &hbuf[wv * 16 * HSTR];

  auto ldfrag = [&](int f) -> v16h {
    const v8h* p = (const v8h*)&wfrag[((f << 5) + nl) << 4];
    return cat8(p[hi], p[32 + hi]);
  };

  for (int gi = 0; gi < 2; ++gi) {
    const int g = wv + gi * 8;

    // A1 fragment: 16x32 f16 from perc (documented A layout -> 2x ds_load_b128)
    const v8h* prow = (const v8h*)&perc[(g * 16 + nl) * PSTR];
    const v16h A1 = cat8(prow[hi], prow[2 + hi]);

    // Layer 1: h1 = relu(A1 * W1eff + b1), 4 N-tiles
    v8f acc[4];
#pragma unroll
    for (int t = 0; t < 4; ++t) {
      const float bb = b1[t * 16 + nl];
      v8f c = { bb, bb, bb, bb, bb, bb, bb, bb };
      acc[t] = wmma16(A1, ldfrag(t), c);
    }
#pragma unroll
    for (int t = 0; t < 4; ++t)
#pragma unroll
      for (int v = 0; v < 8; ++v)
        hb[(v + 8 * hi) * HSTR + t * 16 + nl] =
            (_Float16)fmaxf(acc[t][v], 0.f);
    __builtin_amdgcn_wave_barrier();

    // A2 fragments (K=64 -> 2 chunks)
    const _Float16* hrow = &hb[nl * HSTR];
    v16h A2[2];
#pragma unroll
    for (int kc = 0; kc < 2; ++kc) {
      const v8h* q0 = (const v8h*)(hrow + kc * 32 + hi * 8);
      const v8h* q1 = (const v8h*)(hrow + kc * 32 + 16 + hi * 8);
      A2[kc] = cat8(*q0, *q1);
    }
    __builtin_amdgcn_wave_barrier();

    // Layer 2: h2 = relu(A2 * W2 + b2)
#pragma unroll
    for (int t = 0; t < 4; ++t) {
      const float bb = b2[t * 16 + nl];
      v8f c = { bb, bb, bb, bb, bb, bb, bb, bb };
      c = wmma16(A2[0], ldfrag(4 + t), c);
      acc[t] = wmma16(A2[1], ldfrag(8 + t), c);
    }
#pragma unroll
    for (int t = 0; t < 4; ++t)
#pragma unroll
      for (int v = 0; v < 8; ++v)
        hb[(v + 8 * hi) * HSTR + t * 16 + nl] =
            (_Float16)fmaxf(acc[t][v], 0.f);
    __builtin_amdgcn_wave_barrier();

    // A3 fragments
    v16h A3[2];
#pragma unroll
    for (int kc = 0; kc < 2; ++kc) {
      const v8h* q0 = (const v8h*)(hrow + kc * 32 + hi * 8);
      const v8h* q1 = (const v8h*)(hrow + kc * 32 + 16 + hi * 8);
      A3[kc] = cat8(*q0, *q1);
    }
    __builtin_amdgcn_wave_barrier();

    // Layer 3: update = A3 * W3 (single 16-channel N-tile)
    v8f u = { 0.f, 0.f, 0.f, 0.f, 0.f, 0.f, 0.f, 0.f };
    u = wmma16(A3[0], ldfrag(12), u);
    u = wmma16(A3[1], ldfrag(13), u);

    // x_new = x + update; store out + compact ch3 planes
#pragma unroll
    for (int v = 0; v < 8; ++v) {
      const int pl = g * 16 + v + 8 * hi;     // D layout: M = v + 8*hi
      const int gh = th + (pl >> 4), gw = tw + (pl & 15);
      const size_t gp = (size_t)(bimg * IMH + gh) * IMW + gw;
      const size_t ad = (gp << 4) + nl;       // channel = nl
      const float xv = x[ad];
      const float xn = xv + u[v];
      out[ad] = xn;
      if (nl == 3) { x3p[gp] = xv; xn3p[gp] = xn; }
    }
  }
}

// =====================================================================
// Kernel 2: life mask from compact ch3 planes; zero only dead pixels
// (alive pixels: x_new * 1.0 == x_new, already in out).
// =====================================================================
__global__ __launch_bounds__(256) void nca_life_kernel(
    const float* __restrict__ x3p, const float* __restrict__ xn3p,
    float* __restrict__ out)
{
  const int idx = blockIdx.x * 256 + threadIdx.x;   // pixel index
  const int wq = idx & 255;
  const int hq = (idx >> 8) & 255;
  const int bq = idx >> 16;
  float mpre = -INFINITY, mpost = -INFINITY;
#pragma unroll
  for (int dh = -1; dh <= 1; ++dh) {
    const int h2 = hq + dh;
    if (h2 < 0 || h2 >= IMH) continue;
#pragma unroll
    for (int dw = -1; dw <= 1; ++dw) {
      const int w2 = wq + dw;
      if (w2 < 0 || w2 >= IMW) continue;
      const size_t q = (size_t)(((bq << 8) + h2) << 8) + w2;
      mpre  = fmaxf(mpre,  x3p[q]);
      mpost = fmaxf(mpost, xn3p[q]);
    }
  }
  if (!((mpre > 0.1f) && (mpost > 0.1f))) {
    float* po = out + ((size_t)idx << 4);
    const v4f z = { 0.f, 0.f, 0.f, 0.f };
    *(v4f*)(po + 0)  = z;
    *(v4f*)(po + 4)  = z;
    *(v4f*)(po + 8)  = z;
    *(v4f*)(po + 12) = z;
  }
}

extern "C" void kernel_launch(void* const* d_in, const int* in_sizes, int n_in,
                              void* d_out, int out_size, void* d_ws, size_t ws_size,
                              hipStream_t stream) {
  const float* x  = (const float*)d_in[0];
  const float* w1 = (const float*)d_in[1];
  const float* b1 = (const float*)d_in[2];
  const float* w2 = (const float*)d_in[3];
  const float* b2 = (const float*)d_in[4];
  const float* w3 = (const float*)d_in[5];
  float* out  = (float*)d_out;
  float* x3p  = (float*)d_ws;                 // NPIX floats
  float* xn3p = x3p + NPIX;                   // NPIX floats (needs 8 MB ws total)

  nca_mlp_kernel<<<BATCH * (IMH / TS) * (IMW / TS), 256, 0, stream>>>(
      x, w1, b1, w2, b2, w3, out, x3p, xn3p);
  nca_life_kernel<<<NPIX / 256, 256, 0, stream>>>(x3p, xn3p, out);
}
